// OctreeConvGnRelu_11828339933630
// MI455X (gfx1250) — matrix operand order repaired and checked by reading, not compiled, not run on hardware
//
#include <hip/hip_runtime.h>
#include <hip/hip_bf16.h>

#define NNODES 300000
#define KNEI   27
#define CIN    32
#define COUT   64
#define NTILES (NNODES / 16)     // 18750 (exact)
#define NPAIRS (NTILES / 2)      // 9375  (exact) -> M=32 per wave
#define NFRAG  (KNEI * 4)        // 108 A fragments (k, t)
#define AQ_HALF (NFRAG * 32)     // 3456 uint4 per half-fragment region
#define AQ_TOT  (AQ_HALF * 2)    // 6912 uint4 = 110592 B

typedef __attribute__((ext_vector_type(16))) __bf16 v16bf;
typedef __attribute__((ext_vector_type(8)))  float  v8f;
typedef __attribute__((ext_vector_type(4)))  float  v4f;

// ---------------------------------------------------------------------------
// Prep 1: convert data [N, 32] f32 -> bf16 (row-major), 4 elems per thread.
// ---------------------------------------------------------------------------
__global__ __launch_bounds__(256) void cvt_data_kernel(
    const float* __restrict__ src, __bf16* __restrict__ dst, int n) {
  int i = (blockIdx.x * 256 + threadIdx.x) * 4;
  if (i + 3 < n) {
    float4 v = *(const float4*)(src + i);
    union { __bf16 b[4]; uint2 u; } p;
    p.b[0] = (__bf16)v.x; p.b[1] = (__bf16)v.y;
    p.b[2] = (__bf16)v.z; p.b[3] = (__bf16)v.w;
    *(uint2*)(dst + i) = p.u;
  }
}

// ---------------------------------------------------------------------------
// Prep 2: build bf16 A-operand (weight^T) in WMMA 16-bit A 16x32 lane layout,
// split into two 16B half-fragments (lane stride 16 B -> conflict-free LDS).
//   asw[kt*32 + lane]           = elems 0..7  of lane's fragment
//   asw[AQ_HALF + kt*32 + lane] = elems 8..15 of lane's fragment
// Fragment (k,t): rows M = o = t*16 + (lane&15),
//   lanes 0-15  hold K (input channel c) = 0..15
//   lanes 16-31 hold K                  = 16..31
// ---------------------------------------------------------------------------
__global__ void build_asw_kernel(const float* __restrict__ w,
                                 uint4* __restrict__ asw) {
  int kt   = blockIdx.x;        // 0..107  (k*4 + t)
  int k    = kt >> 2;
  int t    = kt & 3;
  int lane = threadIdx.x;       // 0..31
  int o    = t * 16 + (lane & 15);
  int choff = (lane >= 16) ? 16 : 0;
  union { __bf16 b[16]; uint4 q[2]; } u;
#pragma unroll
  for (int j = 0; j < 16; ++j) {
    int c = j + choff;                         // contiguous K = 16h + j
    u.b[j] = (__bf16)w[(k * CIN + c) * COUT + o];
  }
  asw[kt * 32 + lane]           = u.q[0];
  asw[AQ_HALF + kt * 32 + lane] = u.q[1];
}

// ---------------------------------------------------------------------------
// Main kernel: gather-GEMM via v_wmma_f32_16x16x32_bf16 + GroupNorm(4) + ReLU.
// One wave = 32 nodes x 64 channels: 8 accumulator tiles, each A fragment
// (from LDS) feeds two WMMAs -> DS:WMMA ratio 1:1.
// ---------------------------------------------------------------------------
template <bool CACHED>
__global__ __launch_bounds__(256) void conv_gn_relu_kernel(
    const float* __restrict__ dataF, const int* __restrict__ neigh,
    const __bf16* __restrict__ dataBf, const uint4* __restrict__ aswG,
    const float* __restrict__ gw, const float* __restrict__ gb,
    float* __restrict__ out) {
  __shared__ uint4 Alds[AQ_TOT];               // 110592 B of 320 KB/WGP
  for (int i = threadIdx.x; i < AQ_TOT; i += 256) Alds[i] = aswG[i];
  __syncthreads();

  const int wave  = threadIdx.x >> 5;
  const int lane  = threadIdx.x & 31;
  const int wpair = blockIdx.x * 8 + wave;     // tile-pair id
  if (wpair >= NPAIRS) return;                 // wave-uniform, after barrier
  const int  node0 = wpair * 32 + (lane & 15);
  const int  node1 = node0 + 16;
  const bool hi    = (lane >= 16);
  const uint hioff = hi ? 32u : 0u;            // byte offset of lane's K-window

  const int rb0 = node0 * KNEI;
  const int rb1 = node1 * KNEI;

  v8f acc0 = {}, acc1 = {}, acc2 = {}, acc3 = {};
  v8f acc4 = {}, acc5 = {}, acc6 = {}, acc7 = {};

#pragma unroll
  for (int k = 0; k < KNEI; ++k) {
    const int nb0 = neigh[rb0 + k];
    const int nb1 = neigh[rb1 + k];
    v16bf b0, b1;
    if (CACHED) {
      // 32-bit byte offsets against scalar base -> GVS (saddr+voffset) loads
      b0 = *(const v16bf*)((const char*)dataBf + (((uint)nb0 << 6) | hioff));
      b1 = *(const v16bf*)((const char*)dataBf + (((uint)nb1 << 6) | hioff));
    } else {
      const float* r0 = dataF + nb0 * CIN + (hi ? 16 : 0);
      const float* r1 = dataF + nb1 * CIN + (hi ? 16 : 0);
      float4 f0 = *(const float4*)(r0 + 0), f1 = *(const float4*)(r0 + 4);
      float4 f2 = *(const float4*)(r0 + 8), f3 = *(const float4*)(r0 + 12);
      float4 g0 = *(const float4*)(r1 + 0), g1 = *(const float4*)(r1 + 4);
      float4 g2 = *(const float4*)(r1 + 8), g3 = *(const float4*)(r1 + 12);
      b0[0] = (__bf16)f0.x; b0[1]  = (__bf16)f0.y; b0[2]  = (__bf16)f0.z; b0[3]  = (__bf16)f0.w;
      b0[4] = (__bf16)f1.x; b0[5]  = (__bf16)f1.y; b0[6]  = (__bf16)f1.z; b0[7]  = (__bf16)f1.w;
      b0[8] = (__bf16)f2.x; b0[9]  = (__bf16)f2.y; b0[10] = (__bf16)f2.z; b0[11] = (__bf16)f2.w;
      b0[12] = (__bf16)f3.x; b0[13] = (__bf16)f3.y; b0[14] = (__bf16)f3.z; b0[15] = (__bf16)f3.w;
      b1[0] = (__bf16)g0.x; b1[1]  = (__bf16)g0.y; b1[2]  = (__bf16)g0.z; b1[3]  = (__bf16)g0.w;
      b1[4] = (__bf16)g1.x; b1[5]  = (__bf16)g1.y; b1[6]  = (__bf16)g1.z; b1[7]  = (__bf16)g1.w;
      b1[8] = (__bf16)g2.x; b1[9]  = (__bf16)g2.y; b1[10] = (__bf16)g2.z; b1[11] = (__bf16)g2.w;
      b1[12] = (__bf16)g3.x; b1[13] = (__bf16)g3.y; b1[14] = (__bf16)g3.z; b1[15] = (__bf16)g3.w;
    }
    const int base = k * 128 + lane;           // (k*4)*32 + lane
    union { uint4 q[2]; v16bf v; } A0, A1, A2, A3;
    A0.q[0] = Alds[base +  0]; A0.q[1] = Alds[AQ_HALF + base +  0];
    A1.q[0] = Alds[base + 32]; A1.q[1] = Alds[AQ_HALF + base + 32];
    A2.q[0] = Alds[base + 64]; A2.q[1] = Alds[AQ_HALF + base + 64];
    A3.q[0] = Alds[base + 96]; A3.q[1] = Alds[AQ_HALF + base + 96];
    acc0 = __builtin_amdgcn_wmma_f32_16x16x32_bf16(false, A0.v, false, b0, (short)0, acc0, false, false);
    acc1 = __builtin_amdgcn_wmma_f32_16x16x32_bf16(false, A1.v, false, b0, (short)0, acc1, false, false);
    acc2 = __builtin_amdgcn_wmma_f32_16x16x32_bf16(false, A2.v, false, b0, (short)0, acc2, false, false);
    acc3 = __builtin_amdgcn_wmma_f32_16x16x32_bf16(false, A3.v, false, b0, (short)0, acc3, false, false);
    acc4 = __builtin_amdgcn_wmma_f32_16x16x32_bf16(false, A0.v, false, b1, (short)0, acc4, false, false);
    acc5 = __builtin_amdgcn_wmma_f32_16x16x32_bf16(false, A1.v, false, b1, (short)0, acc5, false, false);
    acc6 = __builtin_amdgcn_wmma_f32_16x16x32_bf16(false, A2.v, false, b1, (short)0, acc6, false, false);
    acc7 = __builtin_amdgcn_wmma_f32_16x16x32_bf16(false, A3.v, false, b1, (short)0, acc7, false, false);
  }

  // --- GroupNorm(groups of 4 channels) + affine + ReLU, fully in-lane ---
  const int obh = hi ? 8 : 0;
  v8f accs[8] = {acc0, acc1, acc2, acc3, acc4, acc5, acc6, acc7};
#pragma unroll
  for (int p = 0; p < 2; ++p) {
    const int node = p ? node1 : node0;
#pragma unroll
    for (int t = 0; t < 4; ++t) {
      const int ob = t * 16 + obh;     // first of 8 consecutive channels
      float x[8];
#pragma unroll
      for (int j = 0; j < 8; ++j) x[j] = accs[p * 4 + t][j];
      float4 w0 = *(const float4*)(gw + ob);
      float4 w1 = *(const float4*)(gw + ob + 4);
      float4 c0 = *(const float4*)(gb + ob);
      float4 c1 = *(const float4*)(gb + ob + 4);
      float y[8];
#pragma unroll
      for (int g = 0; g < 2; ++g) {
        float a0 = x[g * 4 + 0], a1 = x[g * 4 + 1], a2 = x[g * 4 + 2], a3 = x[g * 4 + 3];
        float m  = (a0 + a1 + a2 + a3) * 0.25f;
        float d0 = a0 - m, d1 = a1 - m, d2 = a2 - m, d3 = a3 - m;
        float var = (d0 * d0 + d1 * d1 + d2 * d2 + d3 * d3) * 0.25f;
        float s = rsqrtf(var + 1e-5f);
        const float4& wv = g ? w1 : w0;
        const float4& cv = g ? c1 : c0;
        y[g * 4 + 0] = fmaxf(d0 * s * wv.x + cv.x, 0.0f);
        y[g * 4 + 1] = fmaxf(d1 * s * wv.y + cv.y, 0.0f);
        y[g * 4 + 2] = fmaxf(d2 * s * wv.z + cv.z, 0.0f);
        y[g * 4 + 3] = fmaxf(d3 * s * wv.w + cv.w, 0.0f);
      }
      // Streamed-once output: non-temporal stores, keep L2 for the gather set.
      v4f y0 = {y[0], y[1], y[2], y[3]};
      v4f y1 = {y[4], y[5], y[6], y[7]};
      __builtin_nontemporal_store(y0, (v4f*)(out + node * COUT + ob));
      __builtin_nontemporal_store(y1, (v4f*)(out + node * COUT + ob + 4));
    }
  }
}

// ---------------------------------------------------------------------------
extern "C" void kernel_launch(void* const* d_in, const int* in_sizes, int n_in,
                              void* d_out, int out_size, void* d_ws, size_t ws_size,
                              hipStream_t stream) {
  const float* data   = (const float*)d_in[0];
  const int*   neigh  = (const int*)d_in[1];
  const float* weight = (const float*)d_in[2];
  const float* gw     = (const float*)d_in[3];
  const float* gb     = (const float*)d_in[4];
  float*       out    = (float*)d_out;

  const size_t dataBfBytes = (size_t)NNODES * CIN * sizeof(__bf16);  // 19.2 MB
  const size_t aswBytes    = (size_t)AQ_TOT * sizeof(uint4);         // 110.6 KB
  const bool cached = ws_size >= dataBfBytes + aswBytes;

  __bf16* dataBf = cached ? (__bf16*)d_ws : nullptr;
  uint4*  asw    = cached ? (uint4*)((char*)d_ws + dataBfBytes) : (uint4*)d_ws;

  build_asw_kernel<<<NFRAG, 32, 0, stream>>>(weight, asw);

  const int gemmBlocks = (NPAIRS + 7) / 8;    // 8 waves (tile-pairs) per block
  if (cached) {
    const int nElem = NNODES * CIN;
    cvt_data_kernel<<<(nElem / 4 + 255) / 256, 256, 0, stream>>>(data, dataBf, nElem);
    conv_gn_relu_kernel<true><<<gemmBlocks, 256, 0, stream>>>(
        data, neigh, dataBf, asw, gw, gb, out);
  } else {
    conv_gn_relu_kernel<false><<<gemmBlocks, 256, 0, stream>>>(
        data, neigh, nullptr, asw, gw, gb, out);
  }
}
